// Block_21998822490503
// MI455X (gfx1250) — matrix operand order repaired and checked by reading, not compiled, not run on hardware
//
#include <hip/hip_runtime.h>
#include <hip/hip_bf16.h>
#include <math.h>
#include <stdint.h>

// ---------------------------------------------------------------------------
// CDNA5 (gfx1250) wave32 WMMA transformer block.
//   v_wmma_f32_16x16x32_bf16 for all GEMM-shaped work, async global->LDS
//   (GLOBAL_LOAD_ASYNC_TO_LDS_B128 / ASYNCcnt) for tile staging.
// Fragment layouts used in this file (contiguous k; A and B agree):
//   A (16x32 bf16):  row = lane&15 ; VGPR i   : K = half*16 + 2i..+1
//   B (32x16 bf16):  col = lane&15 ; VGPR i   : K = half*16 + 2i..+1
//   C/D (16x16 f32): col = lane&15 ; VGPR r   : row = r + 8*half
// where half = (lane>=16).
// ---------------------------------------------------------------------------

typedef __attribute__((ext_vector_type(16))) __bf16 v16bf;
typedef __attribute__((ext_vector_type(8)))  float  v8f;

union FragAB {
    v16bf h;
    unsigned u[8];
};

static __device__ inline unsigned short f2bf(float f) {
    // round-to-nearest-even fp32 -> bf16
    unsigned u = __float_as_uint(f);
    u += 0x7fffu + ((u >> 16) & 1u);
    return (unsigned short)(u >> 16);
}

static __device__ inline v8f v8f_zero() {
    v8f z = {0.f, 0.f, 0.f, 0.f, 0.f, 0.f, 0.f, 0.f};
    return z;
}

// Async DMA: global (16B per lane) -> LDS, tracked by ASYNCcnt.
// LDS operand = low 32 bits of the generic pointer (flat->LDS truncation,
// CDNA5 ISA); global address = VGPR pair.
static __device__ inline void async_load_b128(const void* gptr, void* ldsptr) {
    unsigned lds32 = (unsigned)(uintptr_t)ldsptr;
    unsigned long long ga = (unsigned long long)(uintptr_t)gptr;
    asm volatile("global_load_async_to_lds_b128 %0, %1, off"
                 :: "v"(lds32), "v"(ga)
                 : "memory");
}

static __device__ inline void wait_async0() {
    asm volatile("s_wait_asynccnt 0" ::: "memory");
}

// ---------------------------------------------------------------------------
// fp32 -> bf16 cast (vectorized, n % 4 == 0)
// ---------------------------------------------------------------------------
__global__ __launch_bounds__(256) void cast_bf16_kernel(
    const float* __restrict__ in, unsigned short* __restrict__ out, int n) {
    int i = (blockIdx.x * 256 + threadIdx.x) * 4;
    if (i < n) {
        float4 f = *(const float4*)(in + i);
        uint2 p;
        p.x = (unsigned)f2bf(f.x) | ((unsigned)f2bf(f.y) << 16);
        p.y = (unsigned)f2bf(f.z) | ((unsigned)f2bf(f.w) << 16);
        *(uint2*)(out + i) = p;
    }
}

// ---------------------------------------------------------------------------
// fp32 [K,N] -> bf16 [N,K] transpose-cast via LDS tile (K,N multiples of 32)
// ---------------------------------------------------------------------------
__global__ __launch_bounds__(1024) void castT_bf16_kernel(
    const float* __restrict__ in, unsigned short* __restrict__ out, int K, int N) {
    __shared__ float tile[32][33];
    int n0 = blockIdx.x * 32, k0 = blockIdx.y * 32;
    int tx = threadIdx.x, ty = threadIdx.y;
    tile[ty][tx] = in[(size_t)(k0 + ty) * N + (n0 + tx)];
    __syncthreads();
    out[(size_t)(n0 + ty) * K + (k0 + tx)] = f2bf(tile[tx][ty]);
}

// ---------------------------------------------------------------------------
// bf16 WMMA GEMM:  C[M,N] = A[M,K] * Bt[N,K]^T + bias
//   128x128 tile per 256-thread WG (8 waves, 2x4), wave tile 64x32, K-step 32.
//   Double-buffered LDS filled by async DMA; compute overlaps next-tile fill.
//   EPI: 0 = fp32 out, 1 = bf16 out, 2 = exact-GELU then bf16 out.
// ---------------------------------------------------------------------------
#define G_STRIDE 40   // LDS row stride in bf16 elems (32 data + 8 pad): 80B, 16B aligned, bank-clean

template <int EPI>
__global__ __launch_bounds__(256) void gemm_bf16_kernel(
    const unsigned short* __restrict__ A,   // [M,K] bf16
    const unsigned short* __restrict__ Bt,  // [N,K] bf16 (pre-transposed)
    const float* __restrict__ bias,         // [N]
    float* __restrict__ Cf, unsigned short* __restrict__ Cb,
    int M, int N, int K) {
    __shared__ unsigned short lds_a[2][128 * G_STRIDE];
    __shared__ unsigned short lds_b[2][128 * G_STRIDE];

    const int tid  = threadIdx.x;
    const int lane = tid & 31;
    const int wave = tid >> 5;
    const int half = (lane >> 4) & 1;
    const int lr   = lane & 15;
    const int wm   = wave >> 2;  // 0..1
    const int wn   = wave & 3;   // 0..3
    const int gm0  = blockIdx.y * 128;
    const int gn0  = blockIdx.x * 128;

    v8f acc[4][2];
    for (int i = 0; i < 4; i++)
        for (int j = 0; j < 2; j++) acc[i][j] = v8f_zero();

    // async-stage one 128x32 K-slice of A and B into LDS buffer `buf`
    auto stage = [&](int kt, int buf) {
        const int k0 = kt * 32;
        for (int c = tid; c < 512; c += 256) {
            int row = c >> 2, seg = c & 3;
            async_load_b128(A + (size_t)(gm0 + row) * K + k0 + seg * 8,
                            &lds_a[buf][row * G_STRIDE + seg * 8]);
            async_load_b128(Bt + (size_t)(gn0 + row) * K + k0 + seg * 8,
                            &lds_b[buf][row * G_STRIDE + seg * 8]);
        }
    };

    const int nk = K >> 5;
    stage(0, 0);
    wait_async0();
    __syncthreads();

    for (int kt = 0; kt < nk; ++kt) {
        const int cur = kt & 1;
        if (kt + 1 < nk) stage(kt + 1, cur ^ 1);  // overlap fill with compute

        FragAB afr[4], bfr[2];
        for (int ms = 0; ms < 4; ms++) {
            int row = wm * 64 + ms * 16 + lr;
            for (int i = 0; i < 4; i++) {
                afr[ms].u[i]     = *(const unsigned*)&lds_a[cur][row * G_STRIDE + half * 16 + 2 * i];
                afr[ms].u[4 + i] = *(const unsigned*)&lds_a[cur][row * G_STRIDE + half * 16 + 8 + 2 * i];
            }
        }
        for (int ns = 0; ns < 2; ns++) {
            int col = wn * 32 + ns * 16 + lr;
            for (int i = 0; i < 8; i++)
                bfr[ns].u[i] = *(const unsigned*)&lds_b[cur][col * G_STRIDE + half * 16 + 2 * i];
        }
        for (int ms = 0; ms < 4; ms++)
            for (int ns = 0; ns < 2; ns++)
                acc[ms][ns] = __builtin_amdgcn_wmma_f32_16x16x32_bf16(
                    false, afr[ms].h, false, bfr[ns].h, (short)0, acc[ms][ns], false, false);

        wait_async0();     // next tile resident
        __syncthreads();   // all waves done reading `cur` (frag uses precede barrier)
    }

    for (int ms = 0; ms < 4; ms++)
        for (int ns = 0; ns < 2; ns++) {
            int n  = gn0 + wn * 32 + ns * 16 + lr;
            float bn = bias[n];
            for (int r = 0; r < 8; r++) {
                int m   = gm0 + wm * 64 + ms * 16 + half * 8 + r;
                float v = acc[ms][ns][r] + bn;
                if (EPI == 2) v = 0.5f * v * (1.0f + erff(v * 0.70710678118654752f));
                if (EPI == 0) Cf[(size_t)m * N + n] = v;
                else          Cb[(size_t)m * N + n] = f2bf(v);
            }
        }
}

// ---------------------------------------------------------------------------
// Causal flash attention: qkv bf16 [B,T,3C] -> y fp32 [B,T,C]
//   WG = (128 q-rows, one head, one batch). 8 waves x 16 q-rows.
//   Per 64-key block: S = Q*K^T (WMMA), online softmax, O += P*V (WMMA).
//   K tile staged by async DMA; V transposed manually into LDS.
// ---------------------------------------------------------------------------
#define AK_STRIDE 72   // 64 + 8 pad (144B rows, 16B aligned, bank-clean)
#define AP_STRIDE 68   // 64 + 4 pad (136B rows, 4B aligned, bank-clean)

__global__ __launch_bounds__(256) void flash_attn_kernel(
    const unsigned short* __restrict__ qkv, float* __restrict__ y) {
    __shared__ unsigned short lds_k[64 * AK_STRIDE];           // K block [key][d]
    __shared__ unsigned short lds_v[64 * AK_STRIDE];           // V block transposed [d][key]
    __shared__ unsigned short lds_p[8 * 16 * AP_STRIDE];       // per-wave P [row][key]

    const int T = 2048, C3 = 3072;
    const int tid  = threadIdx.x;
    const int lane = tid & 31;
    const int wave = tid >> 5;
    const int half = (lane >> 4) & 1;
    const int lr   = lane & 15;
    const int h    = blockIdx.y;
    const int b    = blockIdx.z;
    const int qb   = blockIdx.x * 128;
    const int q_lo = qb + wave * 16;
    const size_t base = (size_t)b * T * C3;
    const int hoff = h * 64;

    // Q fragments (held in registers for the whole kernel)
    FragAB qf[2];
    {
        const unsigned short* qrow = qkv + base + (size_t)(q_lo + lr) * C3 + hoff;
        for (int f = 0; f < 2; f++)
            for (int i = 0; i < 4; i++) {
                qf[f].u[i]     = *(const unsigned*)(qrow + f * 32 + half * 16 + 2 * i);
                qf[f].u[4 + i] = *(const unsigned*)(qrow + f * 32 + half * 16 + 8 + 2 * i);
            }
    }

    float mrow[8], lrow[8];
    v8f o[4];
    for (int r = 0; r < 8; r++) { mrow[r] = -3.0e38f; lrow[r] = 0.f; }
    for (int nt = 0; nt < 4; nt++) o[nt] = v8f_zero();

    unsigned short* myp = lds_p + wave * 16 * AP_STRIDE;
    const int nkb = qb / 64 + 2;  // key blocks needed for causal coverage of this WG

    for (int kbi = 0; kbi < nkb; kbi++) {
        const int kb = kbi * 64;
        // stage K block [key][d] via async DMA (16B chunks)
        for (int c = tid; c < 512; c += 256) {
            int row = c >> 3, seg = c & 7;
            async_load_b128(qkv + base + (size_t)(kb + row) * C3 + 1024 + hoff + seg * 8,
                            &lds_k[row * AK_STRIDE + seg * 8]);
        }
        // stage V block transposed: lds_v[d][key]
        for (int c = tid; c < 2048; c += 256) {
            int key = c >> 5, dp = (c & 31) * 2;
            unsigned two = *(const unsigned*)(qkv + base + (size_t)(kb + key) * C3 + 2048 + hoff + dp);
            lds_v[dp * AK_STRIDE + key]       = (unsigned short)(two & 0xffffu);
            lds_v[(dp + 1) * AK_STRIDE + key] = (unsigned short)(two >> 16);
        }
        wait_async0();
        __syncthreads();

        // S = (Q K^T) / 8 with causal mask, 4 tiles of 16 keys
        v8f s[4];
        for (int nt = 0; nt < 4; nt++) {
            FragAB kf[2];
            int col = nt * 16 + lr;
            for (int kk = 0; kk < 2; kk++)
                for (int i = 0; i < 8; i++)
                    kf[kk].u[i] = *(const unsigned*)&lds_k[col * AK_STRIDE + kk * 32 + half * 16 + 2 * i];
            v8f z = v8f_zero();
            z = __builtin_amdgcn_wmma_f32_16x16x32_bf16(false, qf[0].h, false, kf[0].h, (short)0, z, false, false);
            z = __builtin_amdgcn_wmma_f32_16x16x32_bf16(false, qf[1].h, false, kf[1].h, (short)0, z, false, false);
            int kcol = kb + nt * 16 + lr;
            for (int r = 0; r < 8; r++) {
                int qr = q_lo + half * 8 + r;
                s[nt][r] = (kcol > qr) ? -3.0e38f : z[r] * 0.125f;
            }
        }
        // online softmax: new row max (16-lane group reduction)
        float mnew[8], fac[8];
        for (int r = 0; r < 8; r++) {
            float v = fmaxf(fmaxf(s[0][r], s[1][r]), fmaxf(s[2][r], s[3][r]));
            for (int msk = 8; msk >= 1; msk >>= 1) v = fmaxf(v, __shfl_xor(v, msk, 16));
            mnew[r] = fmaxf(mrow[r], v);
            fac[r]  = expf(mrow[r] - mnew[r]);
        }
        // P = exp(S - mnew) -> bf16 into per-wave LDS (C-layout -> A-layout bridge)
        for (int nt = 0; nt < 4; nt++)
            for (int r = 0; r < 8; r++) {
                float p  = expf(s[nt][r] - mnew[r]);
                s[nt][r] = p;
                myp[(half * 8 + r) * AP_STRIDE + nt * 16 + lr] = f2bf(p);
            }
        for (int r = 0; r < 8; r++) {
            float rs = (s[0][r] + s[1][r]) + (s[2][r] + s[3][r]);
            for (int msk = 8; msk >= 1; msk >>= 1) rs += __shfl_xor(rs, msk, 16);
            lrow[r] = lrow[r] * fac[r] + rs;
            mrow[r] = mnew[r];
        }
        for (int nt = 0; nt < 4; nt++)
            for (int r = 0; r < 8; r++) o[nt][r] *= fac[r];

        // P fragments (A-layout) -- same-wave LDS RAW, compiler inserts s_wait_dscnt
        FragAB pf[2];
        for (int kk = 0; kk < 2; kk++)
            for (int i = 0; i < 4; i++) {
                pf[kk].u[i]     = *(const unsigned*)&myp[lr * AP_STRIDE + kk * 32 + half * 16 + 2 * i];
                pf[kk].u[4 + i] = *(const unsigned*)&myp[lr * AP_STRIDE + kk * 32 + half * 16 + 8 + 2 * i];
            }
        // O += P * V
        for (int nt = 0; nt < 4; nt++) {
            FragAB vf;
            for (int kk = 0; kk < 2; kk++) {
                for (int i = 0; i < 8; i++)
                    vf.u[i] = *(const unsigned*)&lds_v[(nt * 16 + lr) * AK_STRIDE + kk * 32 + half * 16 + 2 * i];
                o[nt] = __builtin_amdgcn_wmma_f32_16x16x32_bf16(
                    false, pf[kk].h, false, vf.h, (short)0, o[nt], false, false);
            }
        }
        __syncthreads();  // protect K/V tiles before next staging
    }

    for (int nt = 0; nt < 4; nt++)
        for (int r = 0; r < 8; r++) {
            int row = q_lo + half * 8 + r;
            int d   = nt * 16 + lr;
            y[((size_t)b * T + row) * 1024 + hoff + d] = o[nt][r] / lrow[r];
        }
}

// ---------------------------------------------------------------------------
// out = LayerNorm(x + res) * g + b ; fp32 out, optional bf16 side output
// One 256-thread block per row of C=1024.
// ---------------------------------------------------------------------------
__global__ __launch_bounds__(256) void residual_ln_kernel(
    const float* __restrict__ x, const float* __restrict__ res,
    const float* __restrict__ g, const float* __restrict__ bta,
    float* __restrict__ outf, unsigned short* __restrict__ outb) {
    const int C = 1024;
    const int row  = blockIdx.x;
    const int tid  = threadIdx.x;
    const int lane = tid & 31;
    const int wave = tid >> 5;
    __shared__ float red[8];

    const float* xr = x   + (size_t)row * C;
    const float* rr = res + (size_t)row * C;
    float4 a = *(const float4*)(xr + tid * 4);
    float4 c = *(const float4*)(rr + tid * 4);
    float v[4] = {a.x + c.x, a.y + c.y, a.z + c.z, a.w + c.w};

    float s = v[0] + v[1] + v[2] + v[3];
    for (int m = 16; m >= 1; m >>= 1) s += __shfl_xor(s, m, 32);
    if (lane == 0) red[wave] = s;
    __syncthreads();
    float mu = 0.f;
    for (int i = 0; i < 8; i++) mu += red[i];
    mu *= (1.0f / C);
    __syncthreads();

    float s2 = 0.f;
    for (int i = 0; i < 4; i++) { float d = v[i] - mu; s2 += d * d; }
    for (int m = 16; m >= 1; m >>= 1) s2 += __shfl_xor(s2, m, 32);
    if (lane == 0) red[wave] = s2;
    __syncthreads();
    float var = 0.f;
    for (int i = 0; i < 8; i++) var += red[i];
    var *= (1.0f / C);
    float rstd = rsqrtf(var + 1e-5f);

    for (int i = 0; i < 4; i++) {
        int cc = tid * 4 + i;
        float ov = (v[i] - mu) * rstd * g[cc] + bta[cc];
        outf[(size_t)row * C + cc] = ov;
        if (outb) outb[(size_t)row * C + cc] = f2bf(ov);
    }
}

// ---------------------------------------------------------------------------
// host launcher
// ---------------------------------------------------------------------------
extern "C" void kernel_launch(void* const* d_in, const int* in_sizes, int n_in,
                              void* d_out, int out_size, void* d_ws, size_t ws_size,
                              hipStream_t stream) {
    (void)in_sizes; (void)n_in; (void)out_size; (void)ws_size;
    const float* x     = (const float*)d_in[0];
    const float* w_qkv = (const float*)d_in[1];
    const float* b_qkv = (const float*)d_in[2];
    const float* ln1_g = (const float*)d_in[3];
    const float* ln1_b = (const float*)d_in[4];
    const float* w_fc1 = (const float*)d_in[5];
    const float* b_fc1 = (const float*)d_in[6];
    const float* w_fc2 = (const float*)d_in[7];
    const float* b_fc2 = (const float*)d_in[8];
    const float* ln2_g = (const float*)d_in[9];
    const float* ln2_b = (const float*)d_in[10];
    float* out = (float*)d_out;

    const int Bsz = 4, T = 2048, C = 1024;
    const int M = Bsz * T;  // 8192
    char* ws = (char*)d_ws;
    const size_t MB = 1024 * 1024;
    // liveness-based layout (~168 MB total):
    unsigned short* xb   = (unsigned short*)(ws + 0);         // 16MB: x bf16 (dead after QKV gemm)
    unsigned short* wT   = (unsigned short*)(ws + 16 * MB);   //  8MB: transposed weight (serial reuse)
    unsigned short* qkvb = (unsigned short*)(ws + 24 * MB);   // 48MB: qkv bf16 (dead after attn)
    float* yatt          = (float*)(ws + 72 * MB);            // 32MB: attn out (dead after ln1)
    float* x1f           = (float*)(ws + 104 * MB);           // 32MB: ln1 out fp32
    unsigned short* x1b  = xb;                                // 16MB: ln1 out bf16 (reuses xb)
    unsigned short* hb   = (unsigned short*)(ws + 24 * MB);   // 64MB: gelu(fc1) bf16 (reuses qkvb+yatt)
    float* mlpf          = (float*)(ws + 136 * MB);           // 32MB: fc2 out fp32

    // 1) x -> bf16
    cast_bf16_kernel<<<(M * C) / 4 / 256, 256, 0, stream>>>(x, xb, M * C);
    // 2) qkv = x @ w_qkv + b_qkv  (bf16 out)
    castT_bf16_kernel<<<dim3(3 * C / 32, C / 32), dim3(32, 32), 0, stream>>>(w_qkv, wT, C, 3 * C);
    gemm_bf16_kernel<1><<<dim3(3 * C / 128, M / 128), 256, 0, stream>>>(
        xb, wT, b_qkv, nullptr, qkvb, M, 3 * C, C);
    // 3) causal flash attention
    flash_attn_kernel<<<dim3(T / 128, 16, Bsz), 256, 0, stream>>>(qkvb, yatt);
    // 4) x1 = LN1(x + attn)
    residual_ln_kernel<<<M, 256, 0, stream>>>(x, yatt, ln1_g, ln1_b, x1f, x1b);
    // 5) h = gelu(x1 @ w_fc1 + b_fc1)  (bf16 out)
    castT_bf16_kernel<<<dim3(4 * C / 32, C / 32), dim3(32, 32), 0, stream>>>(w_fc1, wT, C, 4 * C);
    gemm_bf16_kernel<2><<<dim3(4 * C / 128, M / 128), 256, 0, stream>>>(
        x1b, wT, b_fc1, nullptr, hb, M, 4 * C, C);
    // 6) mlp = h @ w_fc2 + b_fc2  (fp32 out)
    castT_bf16_kernel<<<dim3(C / 32, 4 * C / 32), dim3(32, 32), 0, stream>>>(w_fc2, wT, 4 * C, C);
    gemm_bf16_kernel<0><<<dim3(C / 128, M / 128), 256, 0, stream>>>(
        hb, wT, b_fc2, mlpf, nullptr, M, C, 4 * C);
    // 7) out = LN2(x1 + mlp)
    residual_ln_kernel<<<M, 256, 0, stream>>>(x1f, mlpf, ln2_g, ln2_b, out, nullptr);
}
